// spectralGCN_outLayer_6004364280508
// MI455X (gfx1250) — matrix operand order repaired and checked by reading, not compiled, not run on hardware
//
#include <hip/hip_runtime.h>
#include <hip/hip_bf16.h>
#include <math.h>

typedef __attribute__((ext_vector_type(16))) _Float16 v16h;
typedef __attribute__((ext_vector_type(8)))  _Float16 v8h;
typedef __attribute__((ext_vector_type(8)))  float    v8f;

// ---------------------------------------------------------------------------
// Optional gfx1250 async global->LDS copy (ASYNCcnt path). Guarded so the
// file still compiles if this toolchain doesn't declare the builtin.
// Probe round 2 showed the builtin exists with params:
//   (int __vector(4) __device__ *src, int __vector(4) __shared__ *dst, Ii, Ii)
// ---------------------------------------------------------------------------
#if defined(__gfx1250__) && __has_builtin(__builtin_amdgcn_global_load_async_to_lds_b128)
#define ASYNC_LDS 1
typedef int v4i_async __attribute__((__vector_size__(16)));
typedef __attribute__((address_space(1))) v4i_async g_v4i;
typedef __attribute__((address_space(3))) v4i_async l_v4i;
__device__ __forceinline__ void async_cp16(const void* g, void* l) {
    // generic->as(1): same 64-bit value; generic->as(3): low 32 bits are the LDS offset
    __builtin_amdgcn_global_load_async_to_lds_b128(
        (g_v4i*)(uintptr_t)g,
        (l_v4i*)(unsigned)(uintptr_t)l, 0, 0);
}
__device__ __forceinline__ void async_wait0() {
#if __has_builtin(__builtin_amdgcn_s_wait_asynccnt)
    __builtin_amdgcn_s_wait_asynccnt(0);
#else
    asm volatile("s_wait_asynccnt 0x0" ::: "memory");
#endif
}
#else
#define ASYNC_LDS 0
#endif

// ---------------------------------------------------------------------------
// Degree / normalization
// ---------------------------------------------------------------------------
__global__ void deg_init_kernel(float* __restrict__ deg, int N) {
    int i = blockIdx.x * blockDim.x + threadIdx.x;
    if (i < N) deg[i] = 1.0f;                       // self-loop contribution
}

__global__ void deg_accum_kernel(const long long* __restrict__ col,
                                 float* __restrict__ deg, int E) {
    int stride = gridDim.x * blockDim.x;
    for (int e = blockIdx.x * blockDim.x + threadIdx.x; e < E; e += stride)
        atomicAdd(&deg[(int)col[e]], 1.0f);
}

__global__ void dinv_kernel(const float* __restrict__ deg,
                            float* __restrict__ dinv, int N) {
    int i = blockIdx.x * blockDim.x + threadIdx.x;
    if (i < N) {
        float d = deg[i];
        dinv[i] = (d > 0.0f) ? rsqrtf(d) : 0.0f;
    }
}

// ---------------------------------------------------------------------------
// Weight pre-pass: B[K,Nc] f32 (row-major) -> Bt[Ncp,K] f16 (transposed,
// zero-padded to Ncp, multiple of 16). Removes all masking/conversion from
// the GEMM inner loop and makes B fragments contiguous 16B loads.
// ---------------------------------------------------------------------------
__global__ void wcvt_kernel(const float* __restrict__ B, _Float16* __restrict__ Bt,
                            int K, int Nc, int Ncp) {
    int idx = blockIdx.x * blockDim.x + threadIdx.x;
    int total = Ncp * K;
    if (idx >= total) return;
    int n = idx / K, k = idx - n * K;
    Bt[(size_t)n * K + k] = (n < Nc) ? (_Float16)B[(size_t)k * Nc + n] : (_Float16)0.0f;
}

// ---------------------------------------------------------------------------
// Dense GEMM: C[M,Nc] = A[M,K] @ B[K,Nc]   (f32 A -> f16, pre-f16 Bt, f32 C)
// 8 waves / 256-thread block; one 16x16 C tile per wave, all waves share the
// same N-tile so the 16-column Bt panel (<=16 KB) is staged in LDS once per
// block (async global->LDS when available). B fragments: two ds_load_b128.
// Requires K % 32 == 0, M % 16 == 0 (true here). EXEC all-1s at every WMMA.
// ---------------------------------------------------------------------------
__global__ void gemm_wmma_kernel(const float* __restrict__ A,
                                 const _Float16* __restrict__ Bt,
                                 float* __restrict__ C,
                                 int M, int K, int Nc, int ldc) {
    extern __shared__ _Float16 sB[];                // [16][K] halves
    const int lane = threadIdx.x & 31;
    const int wave = threadIdx.x >> 5;
    const int tileN = blockIdx.y;

    // ---- stage Bt rows [tileN*16, tileN*16+16) into LDS (16B chunks) ----
    {
        const v8h* src8 = (const v8h*)(Bt + (size_t)tileN * 16 * K);
        v8h* dst8 = (v8h*)sB;
        int chunks = (16 * K) >> 3;                 // #16-byte chunks
        for (int i = threadIdx.x; i < chunks; i += blockDim.x) {
#if ASYNC_LDS
            async_cp16(src8 + i, dst8 + i);
#else
            dst8[i] = src8[i];
#endif
        }
#if ASYNC_LDS
        async_wait0();
#endif
        __syncthreads();
    }

    const int tileM = blockIdx.x * 8 + wave;
    if (tileM >= (M >> 4)) return;                  // wave-uniform exit (after barrier)

    const int mrow = tileM * 16 + (lane & 15);      // A row this lane feeds
    const int ncol = tileN * 16 + (lane & 15);      // C/B column this lane feeds
    const int nl   = lane & 15;                     // local B column in LDS
    const int kbA  = (lane < 16) ? 0 : 8;           // A-fragment K base (16-bit layout)
    const int kbB  = (lane < 16) ? 0 : 8;           // B-fragment K base

    v8f c = {};
#pragma unroll 4
    for (int k0 = 0; k0 < K; k0 += 32) {
        // ---- A fragment: row mrow, K = {k0+kbA+0..7, k0+kbA+16..23}
        const float* ap = A + (size_t)mrow * K + k0 + kbA;
        float4 a0 = *(const float4*)(ap +  0);
        float4 a1 = *(const float4*)(ap +  4);
        float4 a2 = *(const float4*)(ap + 16);
        float4 a3 = *(const float4*)(ap + 20);
        v16h av;
        av[ 0]=(_Float16)a0.x; av[ 1]=(_Float16)a0.y; av[ 2]=(_Float16)a0.z; av[ 3]=(_Float16)a0.w;
        av[ 4]=(_Float16)a1.x; av[ 5]=(_Float16)a1.y; av[ 6]=(_Float16)a1.z; av[ 7]=(_Float16)a1.w;
        av[ 8]=(_Float16)a2.x; av[ 9]=(_Float16)a2.y; av[10]=(_Float16)a2.z; av[11]=(_Float16)a2.w;
        av[12]=(_Float16)a3.x; av[13]=(_Float16)a3.y; av[14]=(_Float16)a3.z; av[15]=(_Float16)a3.w;

        // ---- B fragment: column ncol, K = {k0+kbB+0..7, k0+kbB+16..23}
        const v8h* bp = (const v8h*)(sB + (size_t)nl * K + k0 + kbB);
        v8h b0 = bp[0];
        v8h b1 = bp[2];
        v16h bv = __builtin_shufflevector(b0, b1, 0,1,2,3,4,5,6,7,8,9,10,11,12,13,14,15);

        c = __builtin_amdgcn_wmma_f32_16x16x32_f16(
                /*neg_a=*/false, av, /*neg_b=*/false, bv,
                /*c_mod=*/(short)0, c, /*reuse_a=*/false, /*reuse_b=*/false);
    }

    // ---- store C tile: VGPR r -> M = tileM*16 + r (+8 for upper half-wave)
    const int mbase = tileM * 16 + ((lane < 16) ? 0 : 8);
    if (ncol < Nc) {
#pragma unroll
        for (int r = 0; r < 8; ++r)
            C[(size_t)(mbase + r) * ldc + ncol] = c[r];
    }
}

// ---------------------------------------------------------------------------
// SpMM part 1: self-loop term + bias (full overwrite -> initializes Out)
// ---------------------------------------------------------------------------
__global__ void spmm_self_kernel(const float* __restrict__ Tin,
                                 const float* __restrict__ dinv,
                                 const float* __restrict__ bias,
                                 float* __restrict__ Out,
                                 int N, int F, int ldo) {
    int idx = blockIdx.x * blockDim.x + threadIdx.x;
    int total = N * F;
    if (idx >= total) return;
    int i = idx / F, f = idx - i * F;
    float w = dinv[i];
    Out[(size_t)i * ldo + f] = w * w * Tin[(size_t)i * F + f] + bias[f];
}

// ---------------------------------------------------------------------------
// SpMM part 2: per-edge gather + scatter-add (one wave per edge, grid-stride)
// ---------------------------------------------------------------------------
__global__ void spmm_edge_kernel(const float* __restrict__ Tin,
                                 const long long* __restrict__ row,
                                 const long long* __restrict__ col,
                                 const float* __restrict__ dinv,
                                 float* __restrict__ Out,
                                 int E, int F, int ldo) {
    const int lane   = threadIdx.x & 31;
    const int wave   = (blockIdx.x * blockDim.x + threadIdx.x) >> 5;
    const int nwaves = (gridDim.x * blockDim.x) >> 5;
    for (int e = wave; e < E; e += nwaves) {
        int r = (int)row[e];
        int c = (int)col[e];
        int en = e + nwaves;
        if (en < E)                                   // prefetch next source row
            __builtin_prefetch(Tin + (size_t)((int)row[en]) * F, 0, 1);
        float w = dinv[r] * dinv[c];
        const float* src = Tin + (size_t)r * F;
        float*       dst = Out + (size_t)c * ldo;
        for (int f = lane; f < F; f += 32)
            atomicAdd(&dst[f], w * src[f]);
    }
}

// ---------------------------------------------------------------------------
// ReLU over an N x F sub-block with row stride ld (in place)
// ---------------------------------------------------------------------------
__global__ void relu_kernel(float* __restrict__ P, int N, int F, int ld) {
    int idx = blockIdx.x * blockDim.x + threadIdx.x;
    int total = N * F;
    if (idx >= total) return;
    int i = idx / F, f = idx - i * F;
    float* p = P + (size_t)i * ld + f;
    float v = *p;
    *p = v > 0.0f ? v : 0.0f;
}

// ---------------------------------------------------------------------------
// Row-wise log_softmax: one wave per row, wave32 shuffle reductions
// ---------------------------------------------------------------------------
__global__ void log_softmax_kernel(const float* __restrict__ In,
                                   float* __restrict__ Out, int N, int F) {
    const int lane = threadIdx.x & 31;
    const int rowi = (blockIdx.x * blockDim.x + threadIdx.x) >> 5;
    if (rowi >= N) return;
    const float* in = In + (size_t)rowi * F;
    float mx = -INFINITY;
    for (int f = lane; f < F; f += 32) mx = fmaxf(mx, in[f]);
#pragma unroll
    for (int off = 16; off > 0; off >>= 1) mx = fmaxf(mx, __shfl_xor(mx, off, 32));
    float s = 0.0f;
    for (int f = lane; f < F; f += 32) s += expf(in[f] - mx);
#pragma unroll
    for (int off = 16; off > 0; off >>= 1) s += __shfl_xor(s, off, 32);
    float lse = mx + logf(s);
    float* out = Out + (size_t)rowi * F;
    for (int f = lane; f < F; f += 32) out[f] = in[f] - lse;
}

// ---------------------------------------------------------------------------
// Host-side orchestration
// ---------------------------------------------------------------------------
static inline int cdiv(int a, int b) { return (a + b - 1) / b; }

extern "C" void kernel_launch(void* const* d_in, const int* in_sizes, int n_in,
                              void* d_out, int out_size, void* d_ws, size_t ws_size,
                              hipStream_t stream) {
    const float*     x    = (const float*)d_in[0];       // [N,512]
    const long long* ei   = (const long long*)d_in[1];   // [2,E] int64
    const float*     eig  = (const float*)d_in[2];       // [N,128]
    const float*     W1   = (const float*)d_in[3];       // [512,128]
    const float*     b1   = (const float*)d_in[4];       // [128]
    const float*     W2   = (const float*)d_in[5];       // [128,64]
    const float*     b2   = (const float*)d_in[6];       // [64]
    const float*     linW = (const float*)d_in[7];       // [128,64]
    const float*     W3   = (const float*)d_in[8];       // [128,40]
    const float*     b3   = (const float*)d_in[9];       // [40]
    float*           out  = (float*)d_out;               // [N,40]

    const int N = in_sizes[0] / 512;
    const int E = in_sizes[1] / 2;
    const long long* row = ei;          // edge_index[0] : source
    const long long* col = ei + E;      // edge_index[1] : target

    const int F0 = 512, H1 = 128, H2 = 64, SP = 128, FC = 128, FO = 40, FOP = 48;

    // workspace layout: f32 region (~106 MB) then f16 transposed weights
    float* ws   = (float*)d_ws;
    float* deg  = ws;
    float* dinv = deg  + N;
    float* t1   = dinv + N;                      // [N,128]  x@W1
    float* h1   = t1   + (size_t)N * H1;         // [N,128]  relu(spmm)
    float* t2   = h1   + (size_t)N * H1;         // [N,64]   h1@W2
    float* hc   = t2   + (size_t)N * H2;         // [N,128]  concat(h2, xp)
    float* t3   = hc   + (size_t)N * FC;         // [N,40]   hc@W3
    float* acc3 = t3   + (size_t)N * FO;         // [N,40]   spmm out
    uintptr_t hbase = ((uintptr_t)(acc3 + (size_t)N * FO) + 255) & ~(uintptr_t)255;
    _Float16* W1t = (_Float16*)hbase;            // [128,512]
    _Float16* W2t = W1t + 128 * 512;             // [64,128]
    _Float16* lWt = W2t + 64 * 128;              // [64,128]
    _Float16* W3t = lWt + 64 * 128;              // [48,128] (zero-padded 40->48)
    (void)ws_size; (void)n_in; (void)out_size;

    const int B256 = 256;

    // 0) weight pre-pass: transpose + f32->f16 (+pad)
    wcvt_kernel<<<cdiv(H1 * F0, B256), B256, 0, stream>>>(W1,   W1t, F0, H1, H1);
    wcvt_kernel<<<cdiv(H2 * H1, B256), B256, 0, stream>>>(W2,   W2t, H1, H2, H2);
    wcvt_kernel<<<cdiv(H2 * SP, B256), B256, 0, stream>>>(linW, lWt, SP, H2, H2);
    wcvt_kernel<<<cdiv(FOP * FC, B256), B256, 0, stream>>>(W3,  W3t, FC, FO, FOP);

    // 1) normalization coefficients
    deg_init_kernel<<<cdiv(N, B256), B256, 0, stream>>>(deg, N);
    deg_accum_kernel<<<2048, B256, 0, stream>>>(col, deg, E);
    dinv_kernel<<<cdiv(N, B256), B256, 0, stream>>>(deg, dinv, N);

    // 2) layer 1: t1 = x @ W1 ; h1 = relu(spmm(t1) + b1)
    {
        dim3 g(cdiv(N / 16, 8), H1 / 16);
        gemm_wmma_kernel<<<g, B256, 16 * F0 * 2, stream>>>(x, W1t, t1, N, F0, H1, H1);
    }
    spmm_self_kernel<<<cdiv(N * H1, B256), B256, 0, stream>>>(t1, dinv, b1, h1, N, H1, H1);
    spmm_edge_kernel<<<4096, B256, 0, stream>>>(t1, row, col, dinv, h1, E, H1, H1);
    relu_kernel<<<cdiv(N * H1, B256), B256, 0, stream>>>(h1, N, H1, H1);

    // 3) layer 2: t2 = h1 @ W2 ; hc[:, :64] = relu(spmm(t2) + b2)
    {
        dim3 g(cdiv(N / 16, 8), H2 / 16);
        gemm_wmma_kernel<<<g, B256, 16 * H1 * 2, stream>>>(h1, W2t, t2, N, H1, H2, H2);
    }
    spmm_self_kernel<<<cdiv(N * H2, B256), B256, 0, stream>>>(t2, dinv, b2, hc, N, H2, FC);
    spmm_edge_kernel<<<4096, B256, 0, stream>>>(t2, row, col, dinv, hc, E, H2, FC);
    relu_kernel<<<cdiv(N * H2, B256), B256, 0, stream>>>(hc, N, H2, FC);

    // 4) spectral branch: hc[:, 64:] = eigenvectors @ lin_W
    {
        dim3 g(cdiv(N / 16, 8), H2 / 16);
        gemm_wmma_kernel<<<g, B256, 16 * SP * 2, stream>>>(eig, lWt, hc + H2, N, SP, H2, FC);
    }

    // 5) output layer: t3 = hc @ W3 ; acc3 = spmm(t3) + b3
    {
        dim3 g(cdiv(N / 16, 8), FOP / 16);       // 3 N-tiles; pad columns are zero
        gemm_wmma_kernel<<<g, B256, 16 * FC * 2, stream>>>(hc, W3t, t3, N, FC, FO, FO);
    }
    spmm_self_kernel<<<cdiv(N * FO, B256), B256, 0, stream>>>(t3, dinv, b3, acc3, N, FO, FO);
    spmm_edge_kernel<<<4096, B256, 0, stream>>>(t3, row, col, dinv, acc3, E, FO, FO);

    // 6) log_softmax rows -> d_out
    log_softmax_kernel<<<cdiv(N * 32, B256), B256, 0, stream>>>(acc3, out, N, FO);
}
